// ExplainedVariationLoss_16372415333124
// MI455X (gfx1250) — matrix operand order repaired and checked by reading, not compiled, not run on hardware
//
#include <hip/hip_runtime.h>
#include <hip/hip_bf16.h>
#include <stdint.h>

typedef __attribute__((ext_vector_type(16))) __bf16 v16bf;
typedef __attribute__((ext_vector_type(8)))  float  v8f;

#define B_      16
#define C_      3
#define K_      196
#define HW_     50176
#define RP_     208            // 200 meaningful rows padded to 13*16
#define MT_     64             // macro tile (4x4 WMMA tiles)
#define NMT_    4              // 4*64 = 256 covers 208 (clipped)
#define NPAIR_  10             // upper triangle of 4x4 macro grid
#define SPLIT_  2              // deterministic split-K (separate partials)
#define NCHTOT_ (HW_ / 32)     // 1568 chunks of K=32
#define NCH_    (NCHTOT_ / SPLIT_)   // 784 (even)
#define LSTR_   80             // LDS row stride bytes (16B aligned, bank-spread)

// fp32 -> packed bf16 pair: round-to-nearest (ties away), pack with v_perm_b32.
__device__ __forceinline__ unsigned pk_bf16(float a, float b) {
    unsigned ua = __float_as_uint(a) + 0x8000u;
    unsigned ub = __float_as_uint(b) + 0x8000u;
    // bytes {ub[3],ub[2],ua[3],ua[2]} -> (bf16(a)) | (bf16(b) << 16)
    return __builtin_amdgcn_perm(ub, ua, 0x07060302u);
}

__device__ __forceinline__ uint4 pack8(float4 lo, float4 hi) {
    uint4 q;
    q.x = pk_bf16(lo.x, lo.y);
    q.y = pk_bf16(lo.z, lo.w);
    q.z = pk_bf16(hi.x, hi.y);
    q.w = pk_bf16(hi.z, hi.w);
    return q;
}

// Row r of the augmented per-batch matrix: [sa(196); x(3); ones(1); zeros...]
__device__ __forceinline__ const float* row_ptr(const float* __restrict__ x,
                                                const float* __restrict__ sa,
                                                int b, int r, float* fill) {
    if (r < K_)      return sa + ((size_t)b * K_ + r) * HW_;
    if (r < K_ + C_) return x  + ((size_t)b * C_ + (r - K_)) * HW_;
    *fill = (r == K_ + C_) ? 1.0f : 0.0f;            // row 199 = ones, rest zero
    return nullptr;
}

__device__ __forceinline__ void load8(const float* __restrict__ p, float fill,
                                      int off, float4& lo, float4& hi) {
    if (p) {
        lo = *(const float4*)(p + off);
        hi = *(const float4*)(p + off + 4);
    } else {
        lo = make_float4(fill, fill, fill, fill);
        hi = lo;
    }
}

union Frag { uint4 q[2]; v16bf v; };

__global__ __launch_bounds__(256) void ev_gram_kernel(
    const float* __restrict__ x, const float* __restrict__ sa,
    float* __restrict__ gpart /* [SPLIT][B][RP][RP] */)
{
    // double-buffered staging: [buf][64 rows][32 bf16, 80B stride]
    __shared__ __align__(16) unsigned char ldsA[2][MT_ * LSTR_];
    __shared__ __align__(16) unsigned char ldsB[2][MT_ * LSTR_];

    const int tid = threadIdx.x;
    int bid = blockIdx.x;
    const int s = bid % SPLIT_;  bid /= SPLIT_;
    int pair = bid % NPAIR_;
    const int b = bid / NPAIR_;

    // pair -> (ti,tj), ti<=tj, row lengths 4,3,2,1 (uniform scalar loop)
    int ti = 0, p = pair;
    while (p >= NMT_ - ti) { p -= NMT_ - ti; ++ti; }
    const int tj = ti + p;

    // ---- loader assignment: 4 threads per row, 8 consecutive floats each ----
    const int lrow = tid >> 2;                  // 0..63
    const int cg   = tid & 3;                   // 8-float group
    float fillA = 0.f, fillB = 0.f;
    const float* pA = row_ptr(x, sa, b, ti * MT_ + lrow, &fillA);
    const float* pB = row_ptr(x, sa, b, tj * MT_ + lrow, &fillB);

    // ---- compute assignment: 8 waves in 4x2, each owns a 16x32 output strip ----
    const int wv   = tid >> 5;
    const int lane = tid & 31;
    const int wr   = wv & 3;                    // tile row 0..3
    const int wc   = wv >> 2;                   // tile-col pair 0..1
    const int m    = lane & 15;
    const int half = lane >> 4;

    const int stoff = lrow * LSTR_ + cg * 16;               // staging store offset
    const int arow  = (16 * wr + m) * LSTR_;                // A-frag row base
    const int brow  = (wc * 32 + m) * LSTR_ + half * 32;    // B-frag base

    v8f acc0 = {0.f,0.f,0.f,0.f,0.f,0.f,0.f,0.f};
    v8f acc1 = {0.f,0.f,0.f,0.f,0.f,0.f,0.f,0.f};

    float4 a0, a1, b0, b1;

#define STAGE(bi)                                                            \
    do {                                                                     \
        *(uint4*)(ldsA[bi] + stoff) = pack8(a0, a1);                         \
        *(uint4*)(ldsB[bi] + stoff) = pack8(b0, b1);                         \
    } while (0)

#define FETCH(eoff)                                                          \
    do {                                                                     \
        load8(pA, fillA, (eoff), a0, a1);                                    \
        load8(pB, fillB, (eoff), b0, b1);                                    \
    } while (0)

#define COMPUTE(bi)                                                          \
    do {                                                                     \
        Frag fa, fb0, fb1;                                                   \
        const unsigned char* ar = ldsA[bi] + arow;                           \
        fa.q[0] = *(const uint4*)(ar + half * 32);                           \
        fa.q[1] = *(const uint4*)(ar + half * 32 + 16);                      \
        const unsigned char* br = ldsB[bi] + brow;                           \
        fb0.q[0] = ((const uint4*)br)[0];                                    \
        fb0.q[1] = ((const uint4*)br)[1];                                    \
        fb1.q[0] = *(const uint4*)(br + 16 * LSTR_);                         \
        fb1.q[1] = *(const uint4*)(br + 16 * LSTR_ + 16);                    \
        acc0 = __builtin_amdgcn_wmma_f32_16x16x32_bf16(                      \
                   false, fa.v, false, fb0.v, (short)0, acc0, false, false); \
        acc1 = __builtin_amdgcn_wmma_f32_16x16x32_bf16(                      \
                   false, fa.v, false, fb1.v, (short)0, acc1, false, false); \
    } while (0)

    // software pipeline, double-buffered LDS: 1 barrier per K-chunk
    int off = (s * NCH_) * 32 + cg * 8;
    FETCH(off);                                  // chunk 0 -> regs
    STAGE(0);                                    // chunk 0 -> buf0
    FETCH(off + 32);                             // chunk 1 -> regs
    for (int c = 0; c < NCH_; c += 2) {
        __syncthreads();                         // buf0 = chunk c ready
        STAGE(1);                                // chunk c+1 -> buf1
        if (c + 2 < NCH_) FETCH(off + 64);       // chunk c+2 -> regs
        COMPUTE(0);                              // chunk c
        __syncthreads();                         // buf1 = chunk c+1 ready
        STAGE(0);                                // chunk c+2 -> buf0 (stale on last iter: write-only, harmless)
        if (c + 3 < NCH_) FETCH(off + 96);       // chunk c+3 -> regs
        COMPUTE(1);                              // chunk c+1
        off += 64;
    }
#undef STAGE
#undef FETCH
#undef COMPUTE

    // ---- epilogue: C layout VGPR i -> M = i + 8*half, N = lane&15 ----
    float* gb = gpart + ((size_t)s * B_ + b) * (size_t)(RP_ * RP_);
    const int gj0 = tj * MT_ + 32 * wc + m;
    const int gj1 = gj0 + 16;
#pragma unroll
    for (int i = 0; i < 8; ++i) {
        int gi = ti * MT_ + 16 * wr + i + 8 * half;
        if (gi < RP_) {
            if (gj0 < RP_) gb[gi * RP_ + gj0] = acc0[i];
            if (gj1 < RP_) gb[gi * RP_ + gj1] = acc1[i];
            if (ti != tj) {                      // mirror into lower triangle
                if (gj0 < RP_) gb[gj0 * RP_ + gi] = acc0[i];
                if (gj1 < RP_) gb[gj1 * RP_ + gi] = acc1[i];
            }
        }
    }
}

__global__ __launch_bounds__(256) void ev_reduce_kernel(
    const float* __restrict__ gpart, float* __restrict__ ssr, float* __restrict__ sst)
{
    const int b = blockIdx.x, tid = threadIdx.x;
    const float* g0 = gpart + (size_t)b * (RP_ * RP_);
    const float* g1 = gpart + ((size_t)B_ + b) * (size_t)(RP_ * RP_);
    auto G = [&](int i, int j) { return g0[i * RP_ + j] + g1[i * RP_ + j]; };

    __shared__ float Mm[C_][K_];
    __shared__ float red1[256], red2[256];

    for (int k = tid; k < K_; k += 256) {
        float inv = 1.0f / (G(K_ + C_, k) + 1e-8f);      // row 199 = counts
        Mm[0][k] = G(196, k) * inv;
        Mm[1][k] = G(197, k) * inv;
        Mm[2][k] = G(198, k) * inv;
    }
    __syncthreads();

    float t1 = 0.f;                                       // <M, S>
    for (int k = tid; k < K_; k += 256)
        t1 += Mm[0][k] * G(196, k) + Mm[1][k] * G(197, k) + Mm[2][k] * G(198, k);

    float t2 = 0.f;                                       // sum_c M_c G M_c^T
    for (int j = tid; j < K_; j += 256) {
        float s0 = 0.f, s1 = 0.f, s2 = 0.f;
        for (int k = 0; k < K_; ++k) {
            float gv = G(k, j);
            s0 += Mm[0][k] * gv; s1 += Mm[1][k] * gv; s2 += Mm[2][k] * gv;
        }
        t2 += s0 * Mm[0][j] + s1 * Mm[1][j] + s2 * Mm[2][j];
    }
    red1[tid] = t1; red2[tid] = t2;
    __syncthreads();
    for (int st = 128; st > 0; st >>= 1) {
        if (tid < st) { red1[tid] += red1[tid + st]; red2[tid] += red2[tid + st]; }
        __syncthreads();
    }
    if (tid == 0) {
        float sxx = G(196, 196) + G(197, 197) + G(198, 198);   // sum x^2
        float sx  = G(199, 196) + G(199, 197) + G(199, 198);   // sum x
        float SSR = sxx - 2.f * red1[0] + red2[0];
        float SST = sxx - sx * sx / (float)(C_ * HW_);
        ssr[b] = SSR; sst[b] = SST;
    }
}

__global__ void ev_final_kernel(const float* __restrict__ ssr,
                                const float* __restrict__ sst,
                                float* __restrict__ out)
{
    if (threadIdx.x == 0 && blockIdx.x == 0) {
        float sr = 0.f, st = 0.f;
        for (int b = 0; b < B_; ++b) { sr += ssr[b]; st += sst[b]; }
        out[0] = sr / (st + 1e-8f);
        for (int b = 0; b < B_; ++b) out[1 + b] = 1.0f - ssr[b] / (sst[b] + 1e-8f);
    }
}

extern "C" void kernel_launch(void* const* d_in, const int* in_sizes, int n_in,
                              void* d_out, int out_size, void* d_ws, size_t ws_size,
                              hipStream_t stream) {
    const float* x  = (const float*)d_in[0];   // (16,3,224,224)
    const float* sa = (const float*)d_in[1];   // (16,196,224,224)
    float* gpart = (float*)d_ws;                              // SPLIT*B*RP*RP floats (~5.54 MB)
    float* ssr   = gpart + (size_t)SPLIT_ * B_ * RP_ * RP_;
    float* sst   = ssr + B_;

    ev_gram_kernel<<<dim3(B_ * NPAIR_ * SPLIT_), 256, 0, stream>>>(x, sa, gpart);
    ev_reduce_kernel<<<dim3(B_), 256, 0, stream>>>(gpart, ssr, sst);
    ev_final_kernel<<<dim3(1), 32, 0, stream>>>(ssr, sst, (float*)d_out);
}
